// KanLayer_54631984005415
// MI455X (gfx1250) — matrix operand order repaired and checked by reading, not compiled, not run on hardware
//
#include <hip/hip_runtime.h>
#include <stdint.h>

typedef __attribute__((ext_vector_type(16))) __bf16 v16bf;
typedef __attribute__((ext_vector_type(8)))  float  v8f;

#define BATCH        16384
#define IN_DIM       4096
#define OUT_DIM      64
#define WAVES_PER_WG 8
#define KSPLIT       2
#define WMAT_ELEMS   (IN_DIM * OUT_DIM)     // 262144 bf16 per weight matrix
#define WMAT_BYTES   (WMAT_ELEMS * 2)       // 524288
#define NBUF         3
// LDS tile: [mat][n][k] with row padded 32->48 elements (96B) so the 16
// column-rows hit staggered bank groups (stride 24 banks, rows i and i+8 alias mod 64).
#define LROW         48
#define LMAT         (64 * LROW)            // elements per mat tile
#define LBUFE        (3 * LMAT)             // elements per slab buffer

// ---------------------------------------------------------------------------
// Elementwise KAN math, branchless & compare-free.
// Exact identities (C2 spline => adjacent cubics differ by a perfect cube):
//   b0_inner - b0_outer = (16/3)(a-1/2)^3
//   mid - b1_neg        = -6x^3
//   mid - b1_hi         = -12(x-1/2)^3
// so min/max compositions reproduce the piecewise spline exactly, and the
// relu's also implement the |x|>=1 -> 0 cutoff.
// ---------------------------------------------------------------------------
__device__ __forceinline__ void kan_elem(float x, float& b0, float& b1, float& s) {
    float a   = __builtin_fabsf(x);
    float am1 = a - 1.0f;
    float b0i = __builtin_fmaf(a * am1, 4.0f * a, 0.6666667f);   // 4a^2(a-1)+2/3
    float t   = __builtin_fmaxf(-am1, 0.0f);                     // relu(1-a)
    float b0o = 1.3333334f * (t * t) * t;
    b0 = __builtin_fminf(b0i, b0o);

    float mid = __builtin_fmaf(
                  __builtin_fmaf(__builtin_fmaf(-4.6666665f, x, 2.0f), x, 1.0f),
                  x, 0.16666667f);
    float u   = __builtin_fmaxf(1.0f - x, 0.0f);
    float w   = __builtin_fmaf(11.0f, x, -2.0f);
    float hiv = 0.6666667f * (u * u) * w;
    float p   = __builtin_fmaxf(x + 0.5f, 0.0f);
    float cap = 1.3333334f * (p * p) * p;
    b1 = __builtin_fminf(__builtin_fmaxf(mid, hiv), cap);

    // SiLU via v_exp_f32 + v_rcp_f32 (no IEEE divide sequence)
    s = x * __builtin_amdgcn_rcpf(1.0f + __expf(-x));
}

// ---------------------------------------------------------------------------
// gfx1250 async global->LDS copy (ASYNCcnt-tracked).
// ---------------------------------------------------------------------------
__device__ __forceinline__ void async_b128(uint32_t lds_addr, uint32_t goff,
                                           uint64_t sbase) {
    asm volatile("global_load_async_to_lds_b128 %0, %1, %2"
                 :: "v"(lds_addr), "v"(goff), "s"(sbase) : "memory");
}

// ---------------------------------------------------------------------------
// Kernel 1: fold c*ws and wb into three bf16 weight matrices, layout
// [mat][k/32][n][k%32] (per-slab 4KB blocks, contiguous per mat).
// ---------------------------------------------------------------------------
__global__ void pack_weights(const float* __restrict__ c,
                             const float* __restrict__ ws,
                             const float* __restrict__ wb,
                             __bf16* __restrict__ wpk) {
    int idx = blockIdx.x * blockDim.x + threadIdx.x;
    if (idx >= WMAT_ELEMS) return;
    int i = idx >> 6;      // k (input_dim)
    int j = idx & 63;      // n (out_dim)
    float w  = ws[idx];
    float w0 = c[(size_t)idx * 2 + 0] * w;
    float w1 = c[(size_t)idx * 2 + 1] * w;
    float w2 = wb[idx];
    size_t base = (((size_t)(i >> 5) * OUT_DIM + j) << 5) + (i & 31);
    wpk[0 * WMAT_ELEMS + base] = (__bf16)w0;
    wpk[1 * WMAT_ELEMS + base] = (__bf16)w1;
    wpk[2 * WMAT_ELEMS + base] = (__bf16)w2;
}

__global__ void zero_out(float* __restrict__ out, int n) {
    int i = blockIdx.x * blockDim.x + threadIdx.x;
    if (i < n) out[i] = 0.0f;
}

// ---------------------------------------------------------------------------
// Kernel 2: fused elementwise + triple-GEMM via v_wmma_f32_16x16x32_bf16.
// B tiles: triple-buffered LDS, filled cooperatively by async copies issued
// 2 slabs ahead; one s_wait_asynccnt + one barrier per slab. x (HBM stream)
// software-pipelined one slab ahead in registers. Split-K=2 combined with
// commutative f32 atomics (deterministic).
// ---------------------------------------------------------------------------
__global__ void __launch_bounds__(256)
kan_gemm(const float* __restrict__ x,
         const __bf16* __restrict__ wpk,
         float* __restrict__ out) {
    __shared__ __align__(128) __bf16 lbuf[NBUF][LBUFE];   // 3 x 18KB

    const int tid  = threadIdx.x;
    const int lane = tid & 31;
    const int wid  = tid >> 5;
    const int r    = lane & 15;
    const int hi   = lane >> 4;

    const int rowBase = (blockIdx.x * WAVES_PER_WG + wid) * 16;
    const int kBase   = blockIdx.y * (IN_DIM / KSPLIT);
    const int nslabs  = (IN_DIM / KSPLIT) / 32;
    const int slab0   = kBase >> 5;

    // async copy: thread t moves 16B chunk t of each 4KB [mat][slab] block;
    // LDS destination re-padded to 96B rows: n = t>>2, kchunk = t&3.
    const uint32_t ldsChunk = (uint32_t)((tid >> 2) * (LROW * 2) + (tid & 3) * 16);
    const uint64_t wbase = (uint64_t)(uintptr_t)wpk;
    uint32_t ldsBase[NBUF];
#pragma unroll
    for (int b = 0; b < NBUF; ++b)
        ldsBase[b] = (uint32_t)(size_t)&lbuf[b][0];

    auto issue = [&](int buf, int slab) {
#pragma unroll
        for (int j = 0; j < 3; ++j)
            async_b128(ldsBase[buf] + j * (LMAT * 2) + ldsChunk,
                       (uint32_t)(j * WMAT_BYTES + slab * 4096 + tid * 16),
                       wbase);
    };

    issue(0, slab0);
    issue(1, slab0 + 1);

    v8f acc[4];
#pragma unroll
    for (int t = 0; t < 4; ++t)
#pragma unroll
        for (int v = 0; v < 8; ++v) acc[t][v] = 0.0f;

    const float* xrow = x + (size_t)(rowBase + r) * IN_DIM;

    // A-fragment-exact x preload (lanes 0-15: K 0..15, lanes 16-31:
    // K 16..31).
    float4 cxa0, cxa1, cxb0, cxb1;
    {
        const float4* pA = (const float4*)(xrow + kBase + hi * 16);
        const float4* pB = (const float4*)(xrow + kBase + 8 + hi * 16);
        cxa0 = pA[0]; cxa1 = pA[1];
        cxb0 = pB[0]; cxb1 = pB[1];
    }

    for (int si = 0; si < nslabs; ++si) {
        // own slab-si copies are the oldest; <=3 leaves only slab si+1 pending
        if (si + 1 < nslabs)
            asm volatile("s_wait_asynccnt 0x3" ::: "memory");
        else
            asm volatile("s_wait_asynccnt 0x0" ::: "memory");
        __syncthreads();   // all waves' si data in LDS; prev slab fully consumed

        if (si + 2 < nslabs) issue((si + 2) % NBUF, slab0 + si + 2);

        const int k0 = kBase + si * 32;
        float4 xa0 = cxa0, xa1 = cxa1, xb0 = cxb0, xb1 = cxb1;
        if (si + 1 < nslabs) {   // issue next slab's x loads (HBM latency hidden)
            const float4* pA = (const float4*)(xrow + k0 + 32 + hi * 16);
            const float4* pB = (const float4*)(xrow + k0 + 40 + hi * 16);
            cxa0 = pA[0]; cxa1 = pA[1];
            cxb0 = pB[0]; cxb1 = pB[1];
        }

        // B fragments from LDS (row stride 24 banks)
        const __bf16* lb = &lbuf[si % NBUF][0];
        v16bf bw[12];
#pragma unroll
        for (int mmat = 0; mmat < 3; ++mmat)
#pragma unroll
            for (int nt = 0; nt < 4; ++nt)
                bw[mmat * 4 + nt] = *(const v16bf*)
                    (lb + mmat * LMAT + (nt * 16 + r) * LROW + hi * 16);

        float xs[16] = {xa0.x, xa0.y, xa0.z, xa0.w, xa1.x, xa1.y, xa1.z, xa1.w,
                        xb0.x, xb0.y, xb0.z, xb0.w, xb1.x, xb1.y, xb1.z, xb1.w};
        v16bf afr[3];
#pragma unroll
        for (int e = 0; e < 16; ++e) {
            float b0, b1, s;
            kan_elem(xs[e], b0, b1, s);
            afr[0][e] = (__bf16)b0;
            afr[1][e] = (__bf16)b1;
            afr[2][e] = (__bf16)s;
        }

#pragma unroll
        for (int nt = 0; nt < 4; ++nt)
#pragma unroll
            for (int mmat = 0; mmat < 3; ++mmat)
                acc[nt] = __builtin_amdgcn_wmma_f32_16x16x32_bf16(
                    false, afr[mmat], false, bw[mmat * 4 + nt],
                    (short)0, acc[nt], false, false);
    }

    // C/D layout: VGPR v -> row = v + hi*8, col = r.
#pragma unroll
    for (int nt = 0; nt < 4; ++nt)
#pragma unroll
        for (int v = 0; v < 8; ++v) {
            int row = rowBase + v + hi * 8;
            int col = nt * 16 + r;
            atomicAdd(out + (size_t)row * OUT_DIM + col, acc[nt][v]);
        }
}

// ---------------------------------------------------------------------------
extern "C" void kernel_launch(void* const* d_in, const int* in_sizes, int n_in,
                              void* d_out, int out_size, void* d_ws, size_t ws_size,
                              hipStream_t stream) {
    (void)in_sizes; (void)n_in; (void)out_size; (void)ws_size;
    const float* x  = (const float*)d_in[0];
    const float* c  = (const float*)d_in[1];
    const float* ws = (const float*)d_in[2];
    const float* wb = (const float*)d_in[3];
    float*  out = (float*)d_out;
    __bf16* wpk = (__bf16*)d_ws;   // 3 * 4096 * 64 bf16 = 1.5 MB scratch

    pack_weights<<<(WMAT_ELEMS + 255) / 256, 256, 0, stream>>>(c, ws, wb, wpk);
    zero_out<<<(BATCH * OUT_DIM + 255) / 256, 256, 0, stream>>>(out, BATCH * OUT_DIM);

    dim3 grid(BATCH / (16 * WAVES_PER_WG), KSPLIT);   // (128, 2)
    kan_gemm<<<grid, 256, 0, stream>>>(x, wpk, out);
}
